// KAN_Convolutional_Layer_37220186587781
// MI455X (gfx1250) — compile-verified, matrix operand order checked by reading, not run on hardware
//
#include <hip/hip_runtime.h>

// ---------------------------------------------------------------------------
// KAN 3x3 conv layer for gfx1250 (MI455X), wave32 + WMMA bf16.
//
// Pipeline (3 kernels on one stream):
//   1) prep_act : x(16,32,64,64) fp32 -> A[b,h,w,288] bf16
//                 A = [ relu(x_c) (32) | B-spline bases b_{c,t} (32*8) ]
//   2) pack_w   : [base_weight ; spline_weight*scaler] -> Wpack, pre-swizzled
//                 into the exact v_wmma 16x16x32 B-fragment lane layout.
//   3) kan_gemm : implicit 3x3 conv, one fused GEMM M=61504,N=64,K=2592
//                 with v_wmma_f32_16x16x32_bf16, f32 accumulation.
// ---------------------------------------------------------------------------

typedef __attribute__((ext_vector_type(16))) __bf16 v16bf;
typedef __attribute__((ext_vector_type(8)))  __bf16 v8bf;
typedef __attribute__((ext_vector_type(8)))  float  v8f;

#define KAN_B   16
#define KAN_C   32
#define KAN_H   64
#define KAN_W   64
#define KAN_HO  62
#define KAN_WO  62
#define KAN_OUT 64
#define KAN_NC  8            // grid_size + spline_order = 5 + 3
#define KAN_KPP 288          // per-pixel features: 32 relu + 32*8 spline
#define KAN_KTOT (9 * KAN_KPP)          // 2592
#define KAN_KBLKS (KAN_KTOT / 32)       // 81

#define A_ELEMS   ((size_t)KAN_B * KAN_H * KAN_W * KAN_KPP)   // 18,874,368 bf16
#define A_BYTES   (A_ELEMS * 2)                               // 37,748,736
#define WPACK_OFF A_BYTES                                     // 256-aligned
#define WPACK_ELEMS ((size_t)KAN_KBLKS * 4 * 32 * 16)         // 165,888 bf16

// --------------------------- cubic B-spline basis ---------------------------
// Uniform grid: h = 0.4, knots g[i] = -1 + (i-3)*0.4, i = 0..11.
// Cox-de Boor, fully unrolled; denominators are k*h (uniform grid).
__device__ __forceinline__ void bspline8(float x, float out[8]) {
    const float h = 0.4f;
    float g[12];
#pragma unroll
    for (int i = 0; i < 12; ++i) g[i] = -1.0f + (float)(i - 3) * h;

    float b0[11];
#pragma unroll
    for (int i = 0; i < 11; ++i)
        b0[i] = (x >= g[i] && x < g[i + 1]) ? 1.0f : 0.0f;

    const float r1 = 1.0f / h, r2 = 1.0f / (2.0f * h), r3 = 1.0f / (3.0f * h);
    float b1[10];
#pragma unroll
    for (int i = 0; i < 10; ++i)
        b1[i] = (x - g[i]) * r1 * b0[i] + (g[i + 2] - x) * r1 * b0[i + 1];
    float b2[9];
#pragma unroll
    for (int i = 0; i < 9; ++i)
        b2[i] = (x - g[i]) * r2 * b1[i] + (g[i + 3] - x) * r2 * b1[i + 1];
#pragma unroll
    for (int i = 0; i < 8; ++i)
        out[i] = (x - g[i]) * r3 * b2[i] + (g[i + 4] - x) * r3 * b2[i + 1];
}

// --------------------------- kernel 1: activations --------------------------
// One thread per (pixel, channel). 16*64*64*32 = 2,097,152 threads.
__global__ void kan_prep_act(const float* __restrict__ x,
                             __bf16* __restrict__ A) {
    const int tid = blockIdx.x * blockDim.x + threadIdx.x;
    const int c   = tid & 31;
    const int pix = tid >> 5;                 // b*4096 + h*64 + w
    const int w   = pix & 63;
    const int h   = (pix >> 6) & 63;
    const int b   = pix >> 12;

    const float v = x[(((size_t)b * KAN_C + c) * KAN_H + h) * KAN_W + w];

    __bf16* ap = A + (size_t)pix * KAN_KPP;
    ap[c] = (__bf16)(v > 0.0f ? v : 0.0f);    // relu part, coalesced b16 stores

    float bs[8];
    bspline8(v, bs);
    v8bf pk;
#pragma unroll
    for (int t = 0; t < 8; ++t) pk[t] = (__bf16)bs[t];
    // 16B-aligned contiguous store of the 8 basis values for this channel.
    *(v8bf*)(ap + 32 + c * 8) = pk;
}

// --------------------------- kernel 2: weight pack ---------------------------
// Fused K order: kk = tap*288 + f ; f<32 -> relu channel, f>=32 -> (c,coef).
// Reference feature order inside a patch is (c, i, j): idx = c*9 + tap.
// Packed into B-fragment layout: lanes 0-15 hold K {0..7,16..23} for column
// N=lane; lanes 16-31 hold K {8..15,24..31} for column N=lane-16.
__global__ void kan_pack_w(const float* __restrict__ base_w,
                           const float* __restrict__ spline_w,
                           const float* __restrict__ scaler,
                           __bf16* __restrict__ Wp) {
    const int tid = blockIdx.x * blockDim.x + threadIdx.x;
    if (tid >= KAN_OUT * KAN_KTOT) return;
    const int o  = tid / KAN_KTOT;
    const int kk = tid - o * KAN_KTOT;
    const int tap = kk / KAN_KPP;
    const int f   = kk - tap * KAN_KPP;

    float wv;
    if (f < 32) {
        wv = base_w[o * KAN_KPP + f * 9 + tap];
    } else {
        const int c = (f - 32) >> 3, q = (f - 32) & 7;
        const int idx = o * KAN_KPP + c * 9 + tap;
        wv = spline_w[(size_t)idx * KAN_NC + q] * scaler[idx];
    }

    const int kblk = kk >> 5, kin = kk & 31;
    const int grp  = (kin >> 3) & 1;                     // lane half
    const int e    = (kin & 7) | ((kin >> 4) << 3);      // element 0..15 in lane
    const int lane = (o & 15) + grp * 16;
    const int nt   = o >> 4;
    Wp[(((size_t)kblk * 4 + nt) * 32 + lane) * 16 + e] = (__bf16)wv;
}

// --------------------------- kernel 3: WMMA GEMM -----------------------------
// One wave per 16(M=wo) x 16(N=out) tile; K loop = 9 taps * 9 blocks of 32.
__global__ void __launch_bounds__(256)
kan_gemm_wmma(const __bf16* __restrict__ A,
              const __bf16* __restrict__ Wp,
              float* __restrict__ out) {
    const int lane = threadIdx.x & 31;
    const int gid  = blockIdx.x * 8 + (threadIdx.x >> 5);   // wave-tile id

    const int nt   = gid & 3;
    const int wo0  = ((gid >> 2) & 3) * 16;
    const int rest = gid >> 4;                               // 0..991
    const int ho   = rest % KAN_HO;
    const int b    = rest / KAN_HO;

    const int mrow  = lane & 15;       // A-matrix M = lane%16 in both halves
    const int khalf = lane >> 4;       // 0: K {0..7,16..23}, 1: K {8..15,24..31}

    // B fragments: lane-contiguous 32B blocks, fully coalesced.
    const __bf16* wlane = Wp + ((size_t)nt * 32 + lane) * 16;

    v8f acc = {};
#pragma unroll
    for (int tap = 0; tap < 9; ++tap) {
        const int i = tap / 3, j = tap % 3;
        int w = wo0 + mrow + j;
        if (w > 63) w = 63;            // clamp: rows wo>=62 are never stored
        const __bf16* arow =
            A + (((size_t)b * KAN_H + (ho + i)) * KAN_W + w) * KAN_KPP
              + khalf * 8;
        __builtin_prefetch(arow, 0, 1);   // global_prefetch_b8 into L2/L0

#pragma unroll
        for (int fb = 0; fb < 9; ++fb) {
            // A fragment: two b128 loads per lane (ISA 16-bit A layout).
            const __bf16* ap = arow + fb * 32;
            v8bf a0 = *(const v8bf*)(ap);
            v8bf a1 = *(const v8bf*)(ap + 16);
            v16bf afrag = __builtin_shufflevector(
                a0, a1, 0, 1, 2, 3, 4, 5, 6, 7, 8, 9, 10, 11, 12, 13, 14, 15);

            // B fragment: two coalesced b128 loads from pre-swizzled weights.
            const __bf16* bp = wlane + (size_t)(tap * 9 + fb) * 4 * 32 * 16;
            v8bf b0 = *(const v8bf*)(bp);
            v8bf b1 = *(const v8bf*)(bp + 8);
            v16bf bfrag = __builtin_shufflevector(
                b0, b1, 0, 1, 2, 3, 4, 5, 6, 7, 8, 9, 10, 11, 12, 13, 14, 15);

            acc = __builtin_amdgcn_wmma_f32_16x16x32_bf16(
                false, afrag, false, bfrag, (short)0, acc, false, false);
        }
    }

    // C/D layout: VGPR r -> M = r + (lane>=16 ? 8 : 0), N = lane & 15.
    const int o = nt * 16 + (lane & 15);
    float* orow = out + (((size_t)b * KAN_OUT + o) * KAN_HO + ho) * KAN_WO;
    const int mofs = khalf * 8;
#pragma unroll
    for (int r = 0; r < 8; ++r) {
        const int wo = wo0 + r + mofs;
        if (wo < KAN_WO) orow[wo] = acc[r];
    }
}

// ------------------------------- launcher ------------------------------------
extern "C" void kernel_launch(void* const* d_in, const int* in_sizes, int n_in,
                              void* d_out, int out_size, void* d_ws, size_t ws_size,
                              hipStream_t stream) {
    const float* x        = (const float*)d_in[0];
    const float* base_w   = (const float*)d_in[1];
    const float* spline_w = (const float*)d_in[2];
    const float* scaler   = (const float*)d_in[3];
    float* out = (float*)d_out;

    __bf16* A  = (__bf16*)d_ws;
    __bf16* Wp = (__bf16*)((char*)d_ws + WPACK_OFF);

    // 1) per-pixel relu + B-spline basis tensor (bf16), 9x reuse via L2.
    {
        const int threads = KAN_B * KAN_H * KAN_W * KAN_C;   // 2,097,152
        kan_prep_act<<<threads / 256, 256, 0, stream>>>(x, A);
    }
    // 2) fuse + swizzle weights into WMMA B-fragment layout.
    {
        const int threads = KAN_OUT * KAN_KTOT;              // 165,888
        kan_pack_w<<<(threads + 255) / 256, 256, 0, stream>>>(
            base_w, spline_w, scaler, Wp);
    }
    // 3) implicit-conv GEMM: 15,872 wave-tiles, 8 wave32 per block.
    {
        const int wave_tiles = KAN_B * KAN_HO * 4 * 4;       // 15,872
        kan_gemm_wmma<<<wave_tiles / 8, 256, 0, stream>>>(A, Wp, out);
    }
}
